// DiffJPEG_31903017074891
// MI455X (gfx1250) — matrix operand order repaired and checked by reading, not compiled, not run on hardware
//
#include <hip/hip_runtime.h>
#include <math.h>

typedef __attribute__((ext_vector_type(16))) _Float16 v16h;
typedef __attribute__((ext_vector_type(8)))  float    v8f;
typedef __attribute__((ext_vector_type(4)))  float    v4f;
typedef __attribute__((ext_vector_type(2)))  float    v2f;

#define PI_F 3.14159265358979323846f
#define FACTOR_F 0.1f     // quality 95 -> (200-190)/100
#define INVSQ2 0.70710678118654752f

// per-wave LDS layout (floats); tile row stride 20 (80B -> rows 16B aligned)
#define YB0  0      // Y tile 0
#define YB1  320    // Y tile 1
#define CBF  640    // Cb staging, tiles 0/1 (640 / 960); tile 0 reused as packed chroma
#define CRF  1280   // Cr staging, tiles 0/1 (1280 / 1600)
#define PCK  640    // packed chroma tile (overwrites Cb staging tile 0)
#define WSZ  1920   // floats per wave
#define ZOFF (8 * WSZ)   // block-shared 320-float zero region (rows 16..31 of any tile)

// All per-wave LDS handoffs are intra-wave: DS pipe is in-order per wave, compiler
// inserts dscnt waits for VGPR hazards -> a compiler ordering fence suffices.
__device__ inline void wsync() {
    __asm__ volatile("" ::: "memory");
    __builtin_amdgcn_wave_barrier();
}

// Y_TABLE = _Y.T (row-major [u][v])
__device__ __constant__ float c_yq[64] = {
    16,12,14,14,18,24,49,72,
    11,12,13,17,22,35,64,92,
    10,14,16,22,37,55,78,95,
    16,19,24,29,56,64,87,98,
    24,26,40,51,68,81,103,112,
    40,58,57,87,109,104,121,100,
    51,60,69,80,103,113,120,103,
    61,55,56,62,77,92,101,99};
__device__ __constant__ float c_cq[64] = {
    17,18,24,47,99,99,99,99,
    18,21,26,66,99,99,99,99,
    24,26,56,99,99,99,99,99,
    47,66,99,99,99,99,99,99,
    99,99,99,99,99,99,99,99,
    99,99,99,99,99,99,99,99,
    99,99,99,99,99,99,99,99,
    99,99,99,99,99,99,99,99};

// Chat = diag(C, C), C[x][u] = cos((2x+1)u*pi/16); k in [0,16)
__device__ inline float chat(int k, int m) {
    const float ang = (float)((2 * (k & 7) + 1) * (m & 7)) * (PI_F / 16.f);
    const float c = __cosf(ang);
    const bool nz = (k < 16) && (((k ^ m) & 8) == 0);
    return nz ? c : 0.f;
}

// diff_round on a pair (packed f32 math -> v_pk_fma_f32):
// x - (1/pi) * sum_{n=1..9} ((-1)^{n+1}/n) sin(2*pi*n*x), Chebyshev recurrence.
__device__ inline v2f diff_round2(v2f x) {
    const v2f th = (2.f * PI_F) * x;
    v2f c2, cur;
    c2[0] = 2.f * __cosf(th[0]);  c2[1] = 2.f * __cosf(th[1]);
    cur[0] = __sinf(th[0]);       cur[1] = __sinf(th[1]);
    v2f sp = {0.f, 0.f};
    v2f s  = {0.f, 0.f};
#pragma unroll
    for (int n = 1; n <= 9; ++n) {
        const float coef = ((n & 1) ? 1.f : -1.f) / (float)n;
        s = s + coef * cur;
        const v2f nxt = c2 * cur - sp;
        sp = cur; cur = nxt;
    }
    return x - s * (1.f / PI_F);
}

// Chat^T * M * Chat on the 16x16 f32 tile at L[base] (row stride 20).
// Group-1 lanes source the B operand from the shared zero region: base select is
// hoisted out of the loop -> 16 unconditional ds loads, no per-element cndmask.
__device__ inline v8f dct16(float* __restrict__ L, int base, int lane, v16h aC, v16h bC) {
    const int g = lane >> 4, mn = lane & 15;
    const int bb = (g ? ZOFF : base) + mn;
    v16h bop;
#pragma unroll
    for (int h = 0; h < 16; ++h) bop[h] = (_Float16)L[bb + h * 20];
    v8f t = {};
    t = __builtin_amdgcn_wmma_f32_16x16x32_f16(false, aC, false, bop, (short)0, t, false, false);
    wsync();
#pragma unroll
    for (int r = 0; r < 8; ++r) L[base + (r + 8 * g) * 20 + mn] = t[r];
    wsync();
    const v4f a0 = *(const v4f*)&L[base + mn * 20 + 8 * g];
    const v4f a1 = *(const v4f*)&L[base + mn * 20 + 8 * g + 4];
    v16h aop;
#pragma unroll
    for (int j = 0; j < 4; ++j) { aop[j] = (_Float16)a0[j]; aop[4 + j] = (_Float16)a1[j]; }
#pragma unroll
    for (int h = 8; h < 16; ++h) aop[h] = (_Float16)0.f;
    v8f d = {};
    d = __builtin_amdgcn_wmma_f32_16x16x32_f16(false, aop, false, bC, (short)0, d, false, false);
    return d;
}

// One wave handles a 16(h) x 32(w) pixel region: 2 luma tiles + 1 fully-packed
// chroma tile (quadrants TL=Cb0 TR=Cb1 BL=Cr0 BR=Cr1).
__global__ __launch_bounds__(256) void diffjpeg_kernel(const float* __restrict__ in,
                                                       float* __restrict__ out) {
    __shared__ float lds[8 * WSZ + 320];
    const int lane = threadIdx.x & 31;
    const int wave = threadIdx.x >> 5;
    const int reg  = blockIdx.x * 8 + wave;     // B*512 regions, multiple of 8
    const int b  = reg >> 9;
    const int ty = (reg >> 4) & 31;
    const int tx = reg & 15;
    const int y0 = ty * 16, x0 = tx * 32;

    const int wb = wave * WSZ;
    float* W = &lds[wb];
    const int g = lane >> 4, mn = lane & 15;

    // block-shared zero region (read-only zeros), one real barrier in the kernel
    for (int i = threadIdx.x; i < 320; i += 256) lds[ZOFF + i] = 0.f;
    __syncthreads();

    // Constant DCT operands; A contiguous k = h + 16g (g=1 -> K>=16 -> zero via mask)
    v16h aC, bC;
#pragma unroll
    for (int h = 0; h < 16; ++h) {
        aC[h] = (_Float16)chat(h + 16 * g, mn);       // A = Chat^T : A[M][K]=Chat[K][M]
    }
#pragma unroll
    for (int h = 0; h < 16; ++h) {
        const int kB = (h & 7) + 8 * g + 16 * (h >> 3); // table order; K>=16 -> zero via mask
        bC[h] = (_Float16)chat(kB, mn);               // B = Chat : B[K][N]
    }

    // Per-lane quant coefficients: u = r (unrolled), v = mn & 7 (per lane)
    const int qv = mn & 7;
    const float fv = (qv == 0) ? INVSQ2 : 1.f;
    float qiY[8], qmY[8], qiC[8], qmC[8];
#pragma unroll
    for (int r = 0; r < 8; ++r) {
        const float scr = 0.25f * fv * ((r == 0) ? INVSQ2 : 1.f);
        const float tY = c_yq[r * 8 + qv] * FACTOR_F;
        const float tC = c_cq[r * 8 + qv] * FACTOR_F;
        qiY[r] = scr * __builtin_amdgcn_rcpf(tY);  qmY[r] = scr * tY;
        qiC[r] = scr * __builtin_amdgcn_rcpf(tC);  qmC[r] = scr * tC;
    }

    // ---- Phase A: load 16x32 RGB region, 4 consecutive pixels/lane (b128 I/O) ----
    const size_t HW = 512 * 512;
    const float* pr = in + (size_t)b * 3 * HW;
    const int vrow = lane >> 3;
    const int c4   = (lane & 7) * 4;
    const int av   = (c4 >> 4) * 320 + (c4 & 15);
#pragma unroll
    for (int chunk = 0; chunk < 4; ++chunk) {
        const int row = chunk * 4 + vrow;
        const size_t idx = (size_t)(y0 + row) * 512 + (x0 + c4);
        const v4f R  = *(const v4f*)&pr[idx]          * 255.f;
        const v4f G_ = *(const v4f*)&pr[HW + idx]     * 255.f;
        const v4f B_ = *(const v4f*)&pr[2 * HW + idx] * 255.f;
        const int a = av + row * 20;
        *(v4f*)&W[YB0 + a] =  0.299f    * R + 0.587f    * G_ + 0.114f    * B_ - 128.f;
        *(v4f*)&W[CBF + a] = -0.168736f * R - 0.331264f * G_ + 0.5f      * B_;
        *(v4f*)&W[CRF + a] =  0.5f      * R - 0.418688f * G_ - 0.081312f * B_;
    }
    wsync();

    // ---- Phase B: 2x2 avg-pool; pack Cb0|Cb1 / Cr0|Cr1 into the 4 quadrants ----
    float pv[8];
#pragma unroll
    for (int r = 0; r < 8; ++r) {
        const int m = r + 8 * g, n = mn;
        const int mm = m & 7, nn = n & 7;
        const int base = ((m >= 8) ? CRF : CBF) + ((n >= 8) ? 320 : 0);
        const v2f t0 = *(const v2f*)&W[base + (2 * mm)     * 20 + 2 * nn];
        const v2f t1 = *(const v2f*)&W[base + (2 * mm + 1) * 20 + 2 * nn];
        pv[r] = 0.25f * (t0[0] + t0[1] + t1[0] + t1[1]);
    }
    wsync();
#pragma unroll
    for (int r = 0; r < 8; ++r) W[PCK + (r + 8 * g) * 20 + mn] = pv[r];
    wsync();

    // ---- Luma tile 0 ----
    v8f d0 = dct16(lds, wb + YB0, lane, aC, bC);
#pragma unroll
    for (int r = 0; r < 8; r += 2) {
        const v2f q = diff_round2((v2f){d0[r] * qiY[r], d0[r + 1] * qiY[r + 1]});
        W[YB0 + (r     + 8 * g) * 20 + mn] = q[0] * qmY[r];
        W[YB0 + (r + 1 + 8 * g) * 20 + mn] = q[1] * qmY[r + 1];
    }
    wsync();
    v8f ry0 = dct16(lds, wb + YB0, lane, aC, bC);

    // ---- Luma tile 1 ----
    v8f d1 = dct16(lds, wb + YB1, lane, aC, bC);
#pragma unroll
    for (int r = 0; r < 8; r += 2) {
        const v2f q = diff_round2((v2f){d1[r] * qiY[r], d1[r + 1] * qiY[r + 1]});
        W[YB1 + (r     + 8 * g) * 20 + mn] = q[0] * qmY[r];
        W[YB1 + (r + 1 + 8 * g) * 20 + mn] = q[1] * qmY[r + 1];
    }
    wsync();
    v8f ry1 = dct16(lds, wb + YB1, lane, aC, bC);

    // ---- Chroma (4 blocks, all quadrants useful) ----
    v8f dc = dct16(lds, wb + PCK, lane, aC, bC);
#pragma unroll
    for (int r = 0; r < 8; r += 2) {
        const v2f q = diff_round2((v2f){dc[r] * qiC[r], dc[r + 1] * qiC[r + 1]});
        W[PCK + (r     + 8 * g) * 20 + mn] = q[0] * qmC[r];
        W[PCK + (r + 1 + 8 * g) * 20 + mn] = q[1] * qmC[r + 1];
    }
    wsync();
    v8f rc = dct16(lds, wb + PCK, lane, aC, bC);

    // ---- Stage reconstructed components ----
    wsync();
#pragma unroll
    for (int r = 0; r < 8; ++r) {
        const int a = (r + 8 * g) * 20 + mn;
        W[YB0 + a] = ry0[r] + 128.f;
        W[YB1 + a] = ry1[r] + 128.f;
        W[PCK + a] = rc[r];               // chroma centered (-128 of convert folded)
    }
    wsync();

    // ---- Phase F: NN-upsample chroma, YCbCr->RGB, clamp (v_med3), b128 stores ----
    float* po = out + (size_t)b * 3 * HW;
    const int ccol = ((c4 & 15) >> 1) + 8 * (c4 >> 4);
#pragma unroll
    for (int chunk = 0; chunk < 4; ++chunk) {
        const int row = chunk * 4 + vrow;
        const int crow = row >> 1;
        const v4f Y4 = *(const v4f*)&W[YB0 + av + row * 20];
        const float cb0 = W[PCK + crow * 20 + ccol],       cb1 = W[PCK + crow * 20 + ccol + 1];
        const float cr0 = W[PCK + (8 + crow) * 20 + ccol], cr1 = W[PCK + (8 + crow) * 20 + ccol + 1];
        const v4f Cb = {cb0, cb0, cb1, cb1};
        const v4f Cr = {cr0, cr0, cr1, cr1};
        v4f R  = Y4 + 1.402f * Cr;
        v4f G_ = Y4 - 0.344136f * Cb - 0.714136f * Cr;
        v4f B_ = Y4 + 1.772f * Cb;
#pragma unroll
        for (int j = 0; j < 4; ++j) {
            R[j]  = __builtin_amdgcn_fmed3f(R[j],  0.f, 255.f) * (1.f / 255.f);
            G_[j] = __builtin_amdgcn_fmed3f(G_[j], 0.f, 255.f) * (1.f / 255.f);
            B_[j] = __builtin_amdgcn_fmed3f(B_[j], 0.f, 255.f) * (1.f / 255.f);
        }
        const size_t idx = (size_t)(y0 + row) * 512 + (x0 + c4);
        *(v4f*)&po[idx]          = R;
        *(v4f*)&po[HW + idx]     = G_;
        *(v4f*)&po[2 * HW + idx] = B_;
    }
}

extern "C" void kernel_launch(void* const* d_in, const int* in_sizes, int n_in,
                              void* d_out, int out_size, void* d_ws, size_t ws_size,
                              hipStream_t stream) {
    (void)n_in; (void)out_size; (void)d_ws; (void)ws_size;
    const float* x = (const float*)d_in[0];
    float* y = (float*)d_out;
    const int B = in_sizes[0] / (3 * 512 * 512);
    const int regions = B * (512 / 16) * (512 / 32);  // one wave per 16x32 region
    const int blocks = regions / 8;                   // 8 waves (256 threads) per block
    diffjpeg_kernel<<<blocks, 256, 0, stream>>>(x, y);
}
